// GRU4REC_69286412419116
// MI455X (gfx1250) — matrix operand order repaired and checked by reading, not compiled, not run on hardware
//
#include <hip/hip_runtime.h>
#include <hip/hip_bf16.h>
#include <math.h>

// Problem constants (match the reference)
#define VOC  50000
#define HID  256
#define BAT  512
#define G3H  (3 * HID)   // 768

typedef __attribute__((ext_vector_type(16))) __bf16 v16bf;
typedef __attribute__((ext_vector_type(8)))  __bf16 v8bf;
typedef __attribute__((ext_vector_type(8)))  float  v8f;
typedef __attribute__((ext_vector_type(4)))  float  v4f;

// ---------------------------------------------------------------------------
// f32 -> bf16 convert (for the hidden state feeding the WMMA A operand)
// ---------------------------------------------------------------------------
__global__ void f32_to_bf16_kernel(const float* __restrict__ in,
                                   __bf16* __restrict__ out, int n) {
    int i = blockIdx.x * blockDim.x + threadIdx.x;
    if (i < n) out[i] = (__bf16)in[i];
}

// ---------------------------------------------------------------------------
// Tiled WMMA GEMM:  out[M,N] = act( A[M,K](bf16) * W[N,K](f32->bf16)^T + bias )
// One wave owns a 16 x (16*SUB) output tile.  Per K-step: one big global-load
// clause (2 A b128 + SUB*4 W b128), then straight-line cvt+wmma.  All W
// addresses are direct expressions off the kernel argument (keeps GLOBAL
// address space -> global_load_b128 clauses, not flat_load serialization).
// W streams through registers f32->bf16 (read once per 16-row M tile -> NT).
//
// Fragment packing as used here (16-bit, wave32; A and B share contiguous K):
//   A 16x32: lanes 0-15 row=lane, K 0..15;          lanes 16-31 K 16..31
//   B 32x16: lanes 0-15 col=lane, K 0..15;          lanes 16-31 K 16..31
//   C/D:     lanes 0-15 rows 0..7 (VGPR r = row r), lanes 16-31 rows 8..15
// ---------------------------------------------------------------------------
template <bool APPLY_TANH, int SUB>
__global__ __launch_bounds__(256)
void wmma_gemm_bf16_kernel(const __bf16* __restrict__ A,   // [M x K] bf16
                           const float*  __restrict__ W,   // [N x K] f32
                           const float*  __restrict__ bias,// [N]
                           float*        __restrict__ out, // [M x N]
                           int M, int N, int K,
                           int n_origin, int ntiles) {
    const int lane   = threadIdx.x & 31;
    const int wave   = threadIdx.x >> 5;
    const int mtiles = M >> 4;                 // M is a multiple of 16
    const int t      = blockIdx.x * 8 + wave;  // 8 waves per block
    if (t >= mtiles * ntiles) return;          // wave-uniform exit

    const int m_base = (t % mtiles) << 4;
    const int n_base = n_origin + (t / mtiles) * (16 * SUB);

    const int hi16    = lane >> 4;             // 0: lanes 0-15, 1: lanes 16-31
    const int mrow    = m_base + (lane & 15);
    const int a_khalf = hi16 ? 16 : 0;         // A contiguous-K pattern
    const int ncol    = lane & 15;
    const int b_kbase = hi16 ? 16 : 0;         // B contiguous-K pattern

    const size_t a_off = (size_t)mrow * K + a_khalf;
    size_t w_off[SUB];
#pragma unroll
    for (int s = 0; s < SUB; ++s)
        w_off[s] = (size_t)(n_base + (s << 4) + ncol) * K + b_kbase;

    v8f acc[SUB];
#pragma unroll
    for (int s = 0; s < SUB; ++s) acc[s] = v8f{};

    for (int k0 = 0; k0 < K; k0 += 32) {
        // ---- stage ALL loads first: one clause, staged waits ----
        v8bf alo = *(const v8bf*)(A + a_off + k0);
        v8bf ahi = *(const v8bf*)(A + a_off + k0 + 8);
        v4f wv[SUB][4];
#pragma unroll
        for (int s = 0; s < SUB; ++s) {
            const float* p = W + w_off[s] + k0;
#pragma unroll
            for (int q = 0; q < 4; ++q)
                wv[s][q] = __builtin_nontemporal_load((const v4f*)(p + 4 * q));
        }

        v16bf a = __builtin_shufflevector(alo, ahi,
                      0,1,2,3,4,5,6,7,8,9,10,11,12,13,14,15);
#pragma unroll
        for (int s = 0; s < SUB; ++s) {
            v16bf b;
#pragma unroll
            for (int q = 0; q < 4; ++q) {
#pragma unroll
                for (int e = 0; e < 4; ++e)
                    b[4 * q + e] = (__bf16)wv[s][q][e];
            }
            acc[s] = __builtin_amdgcn_wmma_f32_16x16x32_bf16(
                         false, a, false, b, (short)0, acc[s],
                         false, false);
        }
    }

    // ---- store: tile is fully in-range by construction ----
    const int row0 = m_base + (hi16 ? 8 : 0);
#pragma unroll
    for (int s = 0; s < SUB; ++s) {
        const int   col = n_base + (s << 4) + ncol;
        const float bv  = bias[col];
#pragma unroll
        for (int r = 0; r < 8; ++r) {
            float v = acc[s][r] + bv;
            __builtin_nontemporal_store(APPLY_TANH ? tanhf(v) : v,
                                        out + (size_t)(row0 + r) * N + col);
        }
    }
}

// Launch helper: full 64-wide tiles with SUB=4, remainder subtiles separately.
template <bool APPLY_TANH>
static inline void gemm_launch(const __bf16* A, const float* W,
                               const float* bias, float* out,
                               int M, int N, int K, hipStream_t stream) {
    const int mtiles = M / 16;
    const int full64 = N / 64;
    if (full64 > 0) {
        const int total = mtiles * full64;
        wmma_gemm_bf16_kernel<APPLY_TANH, 4>
            <<<(total + 7) / 8, 256, 0, stream>>>(A, W, bias, out,
                                                  M, N, K, 0, full64);
    }
    const int rem = (N - full64 * 64) / 16;   // leftover 16-wide subtiles
    const int n_origin = full64 * 64;
    if (rem == 1) {
        wmma_gemm_bf16_kernel<APPLY_TANH, 1>
            <<<(mtiles + 7) / 8, 256, 0, stream>>>(A, W, bias, out,
                                                   M, N, K, n_origin, 1);
    } else if (rem == 2) {
        wmma_gemm_bf16_kernel<APPLY_TANH, 2>
            <<<(mtiles + 7) / 8, 256, 0, stream>>>(A, W, bias, out,
                                                   M, N, K, n_origin, 1);
    } else if (rem == 3) {
        wmma_gemm_bf16_kernel<APPLY_TANH, 3>
            <<<(mtiles + 7) / 8, 256, 0, stream>>>(A, W, bias, out,
                                                   M, N, K, n_origin, 1);
    }
}

// ---------------------------------------------------------------------------
// GRU elementwise: gi via one-hot gather from w_ih, gh from the GEMM above
// (bias b_hh already folded in).  Emits f32 h1 (output #2) + bf16 h1 copy
// feeding the logits GEMM.
// ---------------------------------------------------------------------------
__global__ __launch_bounds__(256)
void gru_elem_kernel(const int*   __restrict__ ids,
                     const float* __restrict__ hidden,  // [B,H]
                     const float* __restrict__ w_ih,    // [3H, V]
                     const float* __restrict__ b_ih,    // [3H]
                     const float* __restrict__ gh,      // [B, 3H] (has b_hh)
                     float*       __restrict__ h1_out,  // [B,H] f32
                     __bf16*      __restrict__ h1_bf) { // [B,H] bf16
    const int idx = blockIdx.x * blockDim.x + threadIdx.x;
    if (idx >= BAT * HID) return;
    const int b = idx >> 8;        // HID == 256
    const int j = idx & 255;
    const int id = ids[b];

    const float gir = w_ih[(size_t)(j        ) * VOC + id] + b_ih[j        ];
    const float giz = w_ih[(size_t)(HID   + j) * VOC + id] + b_ih[HID   + j];
    const float gin = w_ih[(size_t)(2*HID + j) * VOC + id] + b_ih[2*HID + j];

    const float ghr = gh[b * G3H + j        ];
    const float ghz = gh[b * G3H + HID   + j];
    const float ghn = gh[b * G3H + 2*HID + j];

    const float r = 1.0f / (1.0f + __expf(-(gir + ghr)));
    const float z = 1.0f / (1.0f + __expf(-(giz + ghz)));
    const float n = tanhf(gin + r * ghn);
    const float h1 = (1.0f - z) * n + z * hidden[idx];

    h1_out[idx] = h1;
    h1_bf[idx]  = (__bf16)h1;
}

// ---------------------------------------------------------------------------
extern "C" void kernel_launch(void* const* d_in, const int* in_sizes, int n_in,
                              void* d_out, int out_size, void* d_ws, size_t ws_size,
                              hipStream_t stream) {
    const int*   ids    = (const int*)  d_in[0];
    const float* hidden = (const float*)d_in[1];
    const float* w_ih   = (const float*)d_in[2];
    const float* w_hh   = (const float*)d_in[3];
    const float* b_ih   = (const float*)d_in[4];
    const float* b_hh   = (const float*)d_in[5];
    const float* w_out  = (const float*)d_in[6];
    const float* b_out  = (const float*)d_in[7];

    float* logits = (float*)d_out;                       // [B, V]
    float* h1_out = logits + (size_t)BAT * VOC;          // [1, B, H]

    // Workspace layout (≈2 MB): h0_bf16 | h1_bf16 | gh_f32
    char*   ws    = (char*)d_ws;
    __bf16* h0_bf = (__bf16*)ws;                                  // B*H bf16
    __bf16* h1_bf = (__bf16*)(ws + (size_t)BAT * HID * 2);        // B*H bf16
    float*  gh    = (float*) (ws + (size_t)BAT * HID * 4);        // B*3H f32

    // 1) hidden -> bf16
    f32_to_bf16_kernel<<<(BAT * HID) / 256, 256, 0, stream>>>(hidden, h0_bf,
                                                              BAT * HID);

    // 2) gh = h0 @ w_hh^T + b_hh   (M=512, N=768, K=256) — no activation
    gemm_launch<false>(h0_bf, w_hh, b_hh, gh, BAT, G3H, HID, stream);

    // 3) GRU elementwise -> h1 (f32 output + bf16 copy)
    gru_elem_kernel<<<(BAT * HID) / 256, 256, 0, stream>>>(
        ids, hidden, w_ih, b_ih, gh, h1_out, h1_bf);

    // 4) logits = tanh(h1 @ w_out^T + b_out)  (M=512, N=50000, K=256)
    gemm_launch<true>(h1_bf, w_out, b_out, logits, BAT, VOC, HID, stream);
}
